// GraphConvolution_56642028700407
// MI455X (gfx1250) — matrix operand order repaired and checked by reading, not compiled, not run on hardware
//
#include <hip/hip_runtime.h>

// ---------------------------------------------------------------------------
// GraphConvolution on MI455X (gfx1250, wave32, WMMA)
//   s        = edge_features @ p^T                       [E]   f32
//   Abf      = bf16(T * s) ; Bbf = bf16(T)               [N,E] (one-time convert)
//   mult     = Abf @ Bbf^T                               [N,N] bf16 WMMA (0.55 TFLOP)
//   adjA     = (eye + (1-eye)*mult) * adj_e              [N,N] fused epilogue
//   HW       = H_v @ weight                              [N,DOUT] f32 WMMA 16x16x4
//   out      = adjA @ HW + bias                          [N,DOUT] f32 WMMA 16x16x4
// Tbf (128 MB) + Abf (128 MB) are L2-resident (192 MB L2) -> GEMM1 compute-bound.
// ---------------------------------------------------------------------------

#define N_NODES 4096
#define N_EDGES 16384
#define DIN     256
#define DOUT    256

typedef __attribute__((ext_vector_type(16))) __bf16 v16bf;
typedef __attribute__((ext_vector_type(8)))  __bf16 v8bf;
typedef __attribute__((ext_vector_type(4)))  __bf16 v4bf;
typedef __attribute__((ext_vector_type(8)))  float  v8f;
typedef __attribute__((ext_vector_type(2)))  float  v2f;

#if __has_builtin(__builtin_amdgcn_wmma_f32_16x16x4_f32)
#define HAVE_WMMA_F32X4 1
#else
#define HAVE_WMMA_F32X4 0
#endif

// ---------------------------------------------------------------- s = ef @ p
__global__ void s_kernel(const float* __restrict__ ef, const float* __restrict__ p,
                         float* __restrict__ s) {
  int e = blockIdx.x * blockDim.x + threadIdx.x;
  if (e < N_EDGES) {
    s[e] = ef[3 * e + 0] * p[0] + ef[3 * e + 1] * p[1] + ef[3 * e + 2] * p[2];
  }
}

// ------------------------------------------- T -> { bf16(T*s), bf16(T) } copy
__global__ void convert_scale_kernel(const float* __restrict__ T,
                                     const float* __restrict__ s,
                                     __bf16* __restrict__ Abf,
                                     __bf16* __restrict__ Bbf) {
  const size_t total = (size_t)N_NODES * N_EDGES;
  size_t i = ((size_t)blockIdx.x * blockDim.x + threadIdx.x) * 4;
  const size_t stride = (size_t)gridDim.x * blockDim.x * 4;
  for (; i < total; i += stride) {
    const float4 v  = *(const float4*)(T + i);
    const float4 sv = *(const float4*)(s + (i & (N_EDGES - 1)));  // e = i % E (row len = E)
    v4bf a, b;
    b[0] = (__bf16)v.x;          b[1] = (__bf16)v.y;
    b[2] = (__bf16)v.z;          b[3] = (__bf16)v.w;
    a[0] = (__bf16)(v.x * sv.x); a[1] = (__bf16)(v.y * sv.y);
    a[2] = (__bf16)(v.z * sv.z); a[3] = (__bf16)(v.w * sv.w);
    *(v4bf*)(Bbf + i) = b;
    *(v4bf*)(Abf + i) = a;
  }
}

// ---------------------------------------------------------------------------
// gemm1: adjA = mask( (T*s) @ T^T , adj_e )
// block = 256 threads = 8 waves arranged 2(M) x 4(N); wave tile = 32x64.
// Block tile = 64(M) x 256(N). Grid = (N/256, N/64).
// A-frag (16-bit 16x32): lane holds row rowBase+ml, K chunks {c0..+7, c0+16..+7},
//   c0 = k + 8*half.  B-frag (32x16): lane holds col colBase+ml = row of T,
//   K chunks {c0..+7, c0+16..+7}, same as A.
// Pure load->WMMA inner loop (A pre-scaled): no VALU between, no hazard NOPs.
// ---------------------------------------------------------------------------
__global__ __launch_bounds__(256) void gemm1_kernel(
    const __bf16* __restrict__ Abf, const __bf16* __restrict__ Bbf,
    const float* __restrict__ adj, float* __restrict__ adjA) {
  const int lane  = threadIdx.x & 31;
  const int wave  = threadIdx.x >> 5;
  const int half  = lane >> 4;
  const int ml    = lane & 15;
  const int waveM = wave >> 2;   // 0..1
  const int waveN = wave & 3;    // 0..3
  const int rowBase = blockIdx.y * 64 + waveM * 32;
  const int colBase = blockIdx.x * 256 + waveN * 64;

  v8f acc[2][4] = {};

  const __bf16* aR[2];
  aR[0] = Abf + (size_t)(rowBase + ml) * N_EDGES;
  aR[1] = aR[0] + (size_t)16 * N_EDGES;
  const __bf16* bR[4];
  bR[0] = Bbf + (size_t)(colBase + ml) * N_EDGES;
  bR[1] = bR[0] + (size_t)16 * N_EDGES;
  bR[2] = bR[0] + (size_t)32 * N_EDGES;
  bR[3] = bR[0] + (size_t)48 * N_EDGES;

#pragma unroll 2
  for (int k = 0; k < N_EDGES; k += 32) {
    const int c0 = k + half * 8;   // A K chunk base (8 + 8 split by +16)
    const int c1 = c0 + 16;
    const int kb = k + half * 16;  // prefetch base; B uses same K chunks as A (c0, c1)

    v16bf a[2], b[4];
#pragma unroll
    for (int i = 0; i < 2; ++i) {
      v8bf lo = *(const v8bf*)(aR[i] + c0);
      v8bf hi = *(const v8bf*)(aR[i] + c1);
      a[i] = __builtin_shufflevector(lo, hi, 0,1,2,3,4,5,6,7,8,9,10,11,12,13,14,15);
    }
#pragma unroll
    for (int j = 0; j < 4; ++j) {
      v8bf lo = *(const v8bf*)(bR[j] + c0);
      v8bf hi = *(const v8bf*)(bR[j] + c1);
      b[j] = __builtin_shufflevector(lo, hi, 0,1,2,3,4,5,6,7,8,9,10,11,12,13,14,15);
    }

    // gfx1250 global_prefetch_b8: pull the K-stream ~4KB ahead (speculative,
    // OOB prefetch past the buffer end is silently dropped).
    __builtin_prefetch(aR[0] + c0 + 2048, 0, 0);
    __builtin_prefetch(bR[0] + kb + 2048, 0, 0);

#pragma unroll
    for (int i = 0; i < 2; ++i)
#pragma unroll
      for (int j = 0; j < 4; ++j)
        acc[i][j] = __builtin_amdgcn_wmma_f32_16x16x32_bf16(
            false, a[i], false, b[j], (short)0, acc[i][j], false, false);
  }

  // Epilogue: M = (row==col) ? 1 : mult ; adjA = M * adj_e
#pragma unroll
  for (int i = 0; i < 2; ++i)
#pragma unroll
    for (int j = 0; j < 4; ++j) {
      const int col = colBase + j * 16 + ml;
#pragma unroll
      for (int r = 0; r < 8; ++r) {
        const int row = rowBase + i * 16 + half * 8 + r;  // C/D: VGPR r -> M = r + 8*half
        float m = acc[i][j][r];
        if (row == col) m = 1.0f;
        adjA[(size_t)row * N_NODES + col] = m * adj[(size_t)row * N_NODES + col];
      }
    }
}

// ---------------------------------------------------------------------------
// hw_kernel: HW = H_v @ weight  (full fp32 via V_WMMA_F32_16X16X4_F32)
// block = 8 waves stacked in M; wave tile = 16(M) x 32(N). Grid = (DOUT/32, N/128).
// ---------------------------------------------------------------------------
__global__ __launch_bounds__(256) void hw_kernel(const float* __restrict__ H,
                                                 const float* __restrict__ W,
                                                 float* __restrict__ HW) {
  const int lane = threadIdx.x & 31;
  const int wave = threadIdx.x >> 5;
  const int half = lane >> 4;
  const int ml   = lane & 15;
  const int rowBase = blockIdx.y * 128 + wave * 16;
  const int colBase = blockIdx.x * 32;
#if HAVE_WMMA_F32X4
  v8f acc0 = {}, acc1 = {};
  const float* aRow = H + (size_t)(rowBase + ml) * DIN;
#pragma unroll 4
  for (int k = 0; k < DIN; k += 4) {
    v2f a = *(const v2f*)(aRow + k + 2 * half);            // A 16x4: lane K = {2h, 2h+1}
    const float* wr = W + (size_t)(k + 2 * half) * DOUT;
    v2f b0, b1;
    b0.x = wr[colBase + ml];       b0.y = wr[DOUT + colBase + ml];
    b1.x = wr[colBase + 16 + ml];  b1.y = wr[DOUT + colBase + 16 + ml];
    acc0 = __builtin_amdgcn_wmma_f32_16x16x4_f32(false, a, false, b0, (short)0, acc0, false, false);
    acc1 = __builtin_amdgcn_wmma_f32_16x16x4_f32(false, a, false, b1, (short)0, acc1, false, false);
  }
#pragma unroll
  for (int r = 0; r < 8; ++r) {
    const int row = rowBase + half * 8 + r;
    HW[(size_t)row * DOUT + colBase + ml]      = acc0[r];
    HW[(size_t)row * DOUT + colBase + 16 + ml] = acc1[r];
  }
#else
  for (int t = threadIdx.x; t < 128 * 32; t += 256) {
    const int row = blockIdx.y * 128 + (t >> 5);
    const int col = blockIdx.x * 32 + (t & 31);
    float s = 0.f;
    for (int k = 0; k < DIN; ++k) s += H[(size_t)row * DIN + k] * W[(size_t)k * DOUT + col];
    HW[(size_t)row * DOUT + col] = s;
  }
#endif
}

// ---------------------------------------------------------------------------
// out_kernel: out = adjA @ HW + bias  (fp32 WMMA, K = 4096)
// ---------------------------------------------------------------------------
__global__ __launch_bounds__(256) void out_kernel(const float* __restrict__ A,
                                                  const float* __restrict__ B,
                                                  const float* __restrict__ bias,
                                                  float* __restrict__ out) {
  const int lane = threadIdx.x & 31;
  const int wave = threadIdx.x >> 5;
  const int half = lane >> 4;
  const int ml   = lane & 15;
  const int rowBase = blockIdx.y * 128 + wave * 16;
  const int colBase = blockIdx.x * 32;
#if HAVE_WMMA_F32X4
  v8f acc0 = {}, acc1 = {};
  const float* aRow = A + (size_t)(rowBase + ml) * N_NODES;
#pragma unroll 4
  for (int k = 0; k < N_NODES; k += 4) {
    v2f a = *(const v2f*)(aRow + k + 2 * half);
    const float* br = B + (size_t)(k + 2 * half) * DOUT;
    v2f b0, b1;
    b0.x = br[colBase + ml];       b0.y = br[DOUT + colBase + ml];
    b1.x = br[colBase + 16 + ml];  b1.y = br[DOUT + colBase + 16 + ml];
    acc0 = __builtin_amdgcn_wmma_f32_16x16x4_f32(false, a, false, b0, (short)0, acc0, false, false);
    acc1 = __builtin_amdgcn_wmma_f32_16x16x4_f32(false, a, false, b1, (short)0, acc1, false, false);
  }
#pragma unroll
  for (int r = 0; r < 8; ++r) {
    const int row = rowBase + half * 8 + r;
    out[(size_t)row * DOUT + colBase + ml]      = acc0[r] + bias[colBase + ml];
    out[(size_t)row * DOUT + colBase + 16 + ml] = acc1[r] + bias[colBase + 16 + ml];
  }
#else
  for (int t = threadIdx.x; t < 128 * 32; t += 256) {
    const int row = blockIdx.y * 128 + (t >> 5);
    const int col = blockIdx.x * 32 + (t & 31);
    float s = bias[col];
    for (int k = 0; k < N_NODES; ++k) s += A[(size_t)row * N_NODES + k] * B[(size_t)k * DOUT + col];
    out[(size_t)row * DOUT + col] = s;
  }
#endif
}

// ---------------------------------------------------------------------------
// Workspace layout (~324 MB)
// ---------------------------------------------------------------------------
static constexpr size_t OFF_A    = 0;
static constexpr size_t SZ_T     = (size_t)N_NODES * N_EDGES * 2;        // 128 MB
static constexpr size_t OFF_B    = OFF_A + SZ_T;                          // 128 MB
static constexpr size_t OFF_S    = OFF_B + SZ_T;
static constexpr size_t SZ_S     = (size_t)N_EDGES * 4;                   // 64 KB f32
static constexpr size_t OFF_ADJA = OFF_S + ((SZ_S + 255) & ~(size_t)255);
static constexpr size_t SZ_ADJA  = (size_t)N_NODES * N_NODES * 4;         // 64 MB
static constexpr size_t OFF_HW   = OFF_ADJA + SZ_ADJA;                    // 4 MB

extern "C" void kernel_launch(void* const* d_in, const int* in_sizes, int n_in,
                              void* d_out, int out_size, void* d_ws, size_t ws_size,
                              hipStream_t stream) {
  (void)in_sizes; (void)n_in; (void)out_size; (void)ws_size;
  const float* H_v  = (const float*)d_in[0];   // [N, DIN]
  const float* ef   = (const float*)d_in[1];   // [E, 3]
  const float* adj  = (const float*)d_in[2];   // [N, N]
  const float* T    = (const float*)d_in[3];   // [N, E]
  const float* W    = (const float*)d_in[4];   // [DIN, DOUT]
  const float* bias = (const float*)d_in[5];   // [DOUT]
  const float* p    = (const float*)d_in[6];   // [3]
  float* out = (float*)d_out;

  char* ws = (char*)d_ws;
  __bf16* Abf  = (__bf16*)(ws + OFF_A);
  __bf16* Bbf  = (__bf16*)(ws + OFF_B);
  float*  sf   = (float*)(ws + OFF_S);
  float*  adjA = (float*)(ws + OFF_ADJA);
  float*  HW   = (float*)(ws + OFF_HW);

  s_kernel<<<N_EDGES / 256, 256, 0, stream>>>(ef, p, sf);
  convert_scale_kernel<<<2048, 256, 0, stream>>>(T, sf, Abf, Bbf);
  hw_kernel<<<dim3(DOUT / 32, N_NODES / 128), 256, 0, stream>>>(H_v, W, HW);
  gemm1_kernel<<<dim3(N_NODES / 256, N_NODES / 64), 256, 0, stream>>>(Abf, Bbf, adj, adjA);
  out_kernel<<<dim3(DOUT / 32, N_NODES / 128), 256, 0, stream>>>(adjA, HW, bias, out);
}
